// S4Block_33028298506608
// MI455X (gfx1250) — compile-verified, matrix operand order checked by reading, not run on hardware
//
#include <hip/hip_runtime.h>
#include <hip/hip_bf16.h>
#include <stdint.h>

// ---- problem constants (match reference) ----
#define Bsz   16
#define Lseq  4096
#define Hdim  256
#define NMODE 32          // N2 = N/2, == wave32 lane count
#define CHUNK 128
#define NCHUNK (Lseq / CHUNK)   // 32

typedef __attribute__((ext_vector_type(16))) _Float16 v16h;
typedef __attribute__((ext_vector_type(8)))  float    v8f;

// gfx1250 async Global->LDS copy (ASYNCcnt-tracked, bypasses VGPRs).
// VDST = per-lane LDS byte address, VADDR = per-lane 64-bit global address.
#define ASYNC_LD_B128(ldsoff, gaddr) \
  asm volatile("global_load_async_to_lds_b128 %0, %1, off" \
               :: "v"(ldsoff), "v"(gaddr) : "memory")
#define ASYNC_LD_B64(ldsoff, gaddr) \
  asm volatile("global_load_async_to_lds_b64 %0, %1, off" \
               :: "v"(ldsoff), "v"(gaddr) : "memory")
#define WAIT_ASYNC(n) asm volatile("s_wait_asynccnt " #n ::: "memory")

// ---------------------------------------------------------------------------
// Kernel 1: per-(h,n) S4D discretization params
//   w  = exp(dtA); Cd = C*(exp(dtA)-1)/A; wL = exp(CHUNK*dtA) = w^128
// params[tid*6] = {w_re, w_im, cd_re, cd_im, wl_re, wl_im}
// ---------------------------------------------------------------------------
__global__ void s4d_params(const float* __restrict__ log_dt,
                           const float* __restrict__ log_A_real,
                           const float* __restrict__ A_imag,
                           const float* __restrict__ C_re,
                           const float* __restrict__ C_im,
                           float* __restrict__ params) {
  int tid = blockIdx.x * blockDim.x + threadIdx.x;
  if (tid >= Hdim * NMODE) return;
  int h = tid / NMODE;
  float dt  = expf(log_dt[h]);
  float Are = -expf(log_A_real[tid]);
  float Aim = A_imag[tid];
  float dre = Are * dt, dim = Aim * dt;

  float e    = expf(dre);
  float w_re = e * cosf(dim), w_im = e * sinf(dim);

  float em_re = w_re - 1.0f, em_im = w_im;
  float inv   = 1.0f / (Are * Are + Aim * Aim);
  float r_re  = (em_re * Are + em_im * Aim) * inv;
  float r_im  = (em_im * Are - em_re * Aim) * inv;
  float cre = C_re[tid], cim = C_im[tid];
  float cd_re = cre * r_re - cim * r_im;
  float cd_im = cre * r_im + cim * r_re;

  float eL    = expf(dre * (float)CHUNK);
  float aL    = dim * (float)CHUNK;
  float wl_re = eL * cosf(aL), wl_im = eL * sinf(aL);

  float* P = params + tid * 6;
  P[0] = w_re; P[1] = w_im; P[2] = cd_re; P[3] = cd_im; P[4] = wl_re; P[5] = wl_im;
}

// ---------------------------------------------------------------------------
// Kernel 2: convert MLP weights to f16 for WMMA
// ---------------------------------------------------------------------------
__global__ void cvt_weights(const float* __restrict__ W1, const float* __restrict__ W2,
                            _Float16* __restrict__ W1h, _Float16* __restrict__ W2h) {
  int tid = blockIdx.x * blockDim.x + threadIdx.x;
  if (tid < Hdim * Hdim) {
    W1h[tid] = (_Float16)W1[tid];
    W2h[tid] = (_Float16)W2[tid];
  }
}

// ---------------------------------------------------------------------------
// Kernel 3: chunked scan pass 1 — per (b,h,chunk) wave, lane = mode.
// ---------------------------------------------------------------------------
__global__ void scan_pass1(const float* __restrict__ x,
                           const float* __restrict__ params,
                           float2* __restrict__ E) {
  int gw   = blockIdx.x * (blockDim.x >> 5) + (threadIdx.x >> 5);
  int lane = threadIdx.x & 31;
  int c = gw & (NCHUNK - 1);
  int h = (gw >> 5) & (Hdim - 1);
  int b = gw >> 13;

  const float* P = params + (h * NMODE + lane) * 6;
  float w_re = P[0], w_im = P[1];
  const float* xp = x + (size_t)(b * Lseq + c * CHUNK) * Hdim + h;

  float s_re = 0.f, s_im = 0.f;
#pragma unroll 4
  for (int l = 0; l < CHUNK; ++l) {
    float u = xp[(size_t)l * Hdim];                  // wave-uniform load
    float nre = fmaf(w_re, s_re, fmaf(-w_im, s_im, u));
    float nim = fmaf(w_re, s_im, w_im * s_re);
    s_re = nre; s_im = nim;
  }
  E[(size_t)gw * NMODE + lane] = make_float2(s_re, s_im);
}

// ---------------------------------------------------------------------------
// Kernel 4: carry combine — per (b,h,n) thread, sequential over 32 chunks.
// ---------------------------------------------------------------------------
__global__ void carry_combine(const float* __restrict__ params,
                              const float2* __restrict__ E,
                              float2* __restrict__ carry) {
  int tid = blockIdx.x * blockDim.x + threadIdx.x;
  if (tid >= Bsz * Hdim * NMODE) return;
  int n  = tid & 31;
  int bh = tid >> 5;
  int h  = bh & (Hdim - 1);
  const float* P = params + (h * NMODE + n) * 6;
  float wl_re = P[4], wl_im = P[5];
  float cr = 0.f, ci = 0.f;
  for (int c = 0; c < NCHUNK; ++c) {
    size_t idx = (size_t)bh * (NCHUNK * NMODE) + c * NMODE + n;
    carry[idx] = make_float2(cr, ci);
    float2 e = E[idx];
    float nr = fmaf(wl_re, cr, fmaf(-wl_im, ci, e.x));
    float ni = fmaf(wl_re, ci, fmaf(wl_im, cr, e.y));
    cr = nr; ci = ni;
  }
}

// ---------------------------------------------------------------------------
// Kernel 5: scan pass 2 — seeded with carries, emits y = 2*Re(Cd·s) + D*u
// ---------------------------------------------------------------------------
__global__ void scan_pass2(const float* __restrict__ x,
                           const float* __restrict__ params,
                           const float2* __restrict__ carry,
                           const float* __restrict__ Dp,
                           float* __restrict__ yconv) {
  int gw   = blockIdx.x * (blockDim.x >> 5) + (threadIdx.x >> 5);
  int lane = threadIdx.x & 31;
  int c = gw & (NCHUNK - 1);
  int h = (gw >> 5) & (Hdim - 1);
  int b = gw >> 13;

  const float* P = params + (h * NMODE + lane) * 6;
  float w_re = P[0], w_im = P[1], cd_re = P[2], cd_im = P[3];
  float dskip = Dp[h];

  float2 s0 = carry[(size_t)gw * NMODE + lane];
  float s_re = s0.x, s_im = s0.y;

  size_t off = (size_t)(b * Lseq + c * CHUNK) * Hdim + h;
  const float* xp = x + off;
  float* yp = yconv + off;

  for (int l = 0; l < CHUNK; ++l) {
    float u = xp[(size_t)l * Hdim];
    float nre = fmaf(w_re, s_re, fmaf(-w_im, s_im, u));
    float nim = fmaf(w_re, s_im, w_im * s_re);
    s_re = nre; s_im = nim;
    float t = fmaf(cd_re, s_re, -cd_im * s_im);
#pragma unroll
    for (int o = 16; o > 0; o >>= 1) t += __shfl_xor(t, o, 32);
    if (lane == 0) yp[(size_t)l * Hdim] = fmaf(dskip, u, 2.0f * t);
  }
}

// ---------------------------------------------------------------------------
// Kernel 6: LayerNorm over H per (b,l) row; one wave per row; f16 output.
// ---------------------------------------------------------------------------
__global__ void layernorm_f16(const float* __restrict__ yconv,
                              const float* __restrict__ g,
                              const float* __restrict__ beta,
                              _Float16* __restrict__ yln) {
  int row  = blockIdx.x * (blockDim.x >> 5) + (threadIdx.x >> 5);
  int lane = threadIdx.x & 31;
  const float* rp = yconv + (size_t)row * Hdim;
  float v[8], sum = 0.f, ssq = 0.f;
#pragma unroll
  for (int i = 0; i < 8; ++i) {
    v[i] = rp[lane + 32 * i];
    sum += v[i];
    ssq = fmaf(v[i], v[i], ssq);
  }
#pragma unroll
  for (int o = 16; o > 0; o >>= 1) {
    sum += __shfl_xor(sum, o, 32);
    ssq += __shfl_xor(ssq, o, 32);
  }
  float mean = sum * (1.0f / Hdim);
  float var  = ssq * (1.0f / Hdim) - mean * mean;
  float rstd = rsqrtf(var + 1e-5f);
#pragma unroll
  for (int i = 0; i < 8; ++i) {
    int hh = lane + 32 * i;
    yln[(size_t)row * Hdim + hh] = (_Float16)(fmaf((v[i] - mean) * rstd, g[hh], beta[hh]));
  }
}

// ---------------------------------------------------------------------------
// Kernel 7: GEMM  out[m,o] = relu( sum_h A[m,h]*W[o,h] + b[o] ) (+ resid)
// 256 threads = 8 waves; block tile 64(M) x 32(N); wave tile 16x16.
// Double-buffered LDS staged via async Global->LDS (ASYNCcnt); K-loop fully
// unrolled with the last tile peeled so the body is branch-free.
// ---------------------------------------------------------------------------
template <bool RESID>
__global__ __launch_bounds__(256)
void gemm_relu(const _Float16* __restrict__ A,
               const _Float16* __restrict__ Wh,
               const float* __restrict__ bias,
               const float* __restrict__ resid,
               _Float16* __restrict__ outh,
               float* __restrict__ outf) {
  constexpr int K   = Hdim;               // 256
  constexpr int NT  = K / 32;             // 8 K-tiles
  constexpr int LDA = 20;                 // padded row stride in u32 (80B)
  __shared__ __align__(16) uint32_t As[2][64 * LDA];  // 64 rows x 16 u32 used
  __shared__ __align__(16) uint32_t Bs[2][32 * LDA];  // 32 rows x 16 u32 used

  int t = threadIdx.x;
  int lane = t & 31, wid = t >> 5;
  int wm = wid >> 1, wn = wid & 1;        // 4x2 wave grid
  int m0 = blockIdx.x * 64;
  int n0 = blockIdx.y * 32;
  int g  = lane >> 4;
  int mn = lane & 15;

  const uint32_t* Ag = (const uint32_t*)A;   // K/2 = 128 u32 per row
  const uint32_t* Wg = (const uint32_t*)Wh;

  // per-thread staging slots
  int arow = t >> 2, acol = (t & 3) * 4;     // A: one b128 (4 u32) per thread
  int brow = t >> 3, bcol = (t & 7) * 2;     // B: one b64  (2 u32) per thread
  const uint32_t* agp = Ag + (size_t)(m0 + arow) * (K / 2) + acol;
  const uint32_t* bgp = Wg + (size_t)(n0 + brow) * (K / 2) + bcol;
  uint32_t alds[2] = { (uint32_t)(uintptr_t)&As[0][arow * LDA + acol],
                       (uint32_t)(uintptr_t)&As[1][arow * LDA + acol] };
  uint32_t blds[2] = { (uint32_t)(uintptr_t)&Bs[0][brow * LDA + bcol],
                       (uint32_t)(uintptr_t)&Bs[1][brow * LDA + bcol] };

  // prologue: async-stage tile 0
  ASYNC_LD_B128(alds[0], agp);
  ASYNC_LD_B64 (blds[0], bgp);

  v8f acc = {};
  const int afrag = (wm * 16 + mn) * LDA;
  const int bfrag = (wn * 16 + mn) * LDA;

  auto mma_tile = [&](int cur) {
    union { v16h v; uint32_t u[8]; } af, bf;
#pragma unroll
    for (int q = 0; q < 8; ++q) {
      // ISA 16-bit fragment layout: K pair base: q<4 -> 2q ; q>=4 -> 16+2(q-4);
      // lanes 16..31 shifted by +8.  -> two contiguous 4-u32 runs per fragment.
      int kb = ((q < 4) ? 2 * q : 16 + 2 * (q - 4)) + 8 * g;
      af.u[q] = As[cur][afrag + (kb >> 1)];
      bf.u[q] = Bs[cur][bfrag + (kb >> 1)];
    }
    acc = __builtin_amdgcn_wmma_f32_16x16x32_f16(false, af.v, false, bf.v,
                                                 (short)0, acc, false, false);
  };

#pragma unroll
  for (int kt = 0; kt < NT - 1; ++kt) {      // branch-free steady state
    int cur = kt & 1;
    ASYNC_LD_B128(alds[cur ^ 1], agp + (kt + 1) * 16);
    ASYNC_LD_B64 (blds[cur ^ 1], bgp + (kt + 1) * 16);
    WAIT_ASYNC(0x2);                         // current tile's 2 ops retired
    __syncthreads();                         // tile visible to all waves
    mma_tile(cur);
    __syncthreads();                         // reads done before buffer reuse
  }
  // peeled last tile: nothing left to prefetch
  WAIT_ASYNC(0x0);
  __syncthreads();
  mma_tile((NT - 1) & 1);

  int ncol = n0 + wn * 16 + mn;
  float bv = bias[ncol];
#pragma unroll
  for (int r = 0; r < 8; ++r) {
    int m = m0 + wm * 16 + r + 8 * g;
    float val = acc[r] + bv;
    val = val > 0.f ? val : 0.f;
    size_t oidx = (size_t)m * Hdim + ncol;
    if (RESID) outf[oidx] = resid[oidx] + val;
    else       outh[oidx] = (_Float16)val;
  }
}

// ---------------------------------------------------------------------------
// Launcher
// ---------------------------------------------------------------------------
extern "C" void kernel_launch(void* const* d_in, const int* in_sizes, int n_in,
                              void* d_out, int out_size, void* d_ws, size_t ws_size,
                              hipStream_t stream) {
  (void)in_sizes; (void)n_in; (void)out_size; (void)ws_size;
  const float* x      = (const float*)d_in[0];
  const float* log_dt = (const float*)d_in[1];
  const float* log_A  = (const float*)d_in[2];
  const float* A_im   = (const float*)d_in[3];
  const float* C_re   = (const float*)d_in[4];
  const float* C_im   = (const float*)d_in[5];
  const float* Dp     = (const float*)d_in[6];
  const float* ln_g   = (const float*)d_in[7];
  const float* ln_b   = (const float*)d_in[8];
  const float* W1     = (const float*)d_in[9];
  const float* b1     = (const float*)d_in[10];
  const float* W2     = (const float*)d_in[11];
  const float* b2     = (const float*)d_in[12];
  float* out = (float*)d_out;

  char* w = (char*)d_ws;
  auto alloc = [&](size_t bytes) -> char* {
    char* p = w; w += (bytes + 255) & ~(size_t)255; return p;
  };
  float*    params = (float*)   alloc((size_t)Hdim * NMODE * 6 * sizeof(float));
  float2*   E      = (float2*)  alloc((size_t)Bsz * Hdim * NCHUNK * NMODE * sizeof(float2));
  float2*   carry  = (float2*)  alloc((size_t)Bsz * Hdim * NCHUNK * NMODE * sizeof(float2));
  float*    yconv  = (float*)   alloc((size_t)Bsz * Lseq * Hdim * sizeof(float));
  _Float16* yln    = (_Float16*)alloc((size_t)Bsz * Lseq * Hdim * sizeof(_Float16));
  _Float16* h1     = (_Float16*)alloc((size_t)Bsz * Lseq * Hdim * sizeof(_Float16));
  _Float16* W1h    = (_Float16*)alloc((size_t)Hdim * Hdim * sizeof(_Float16));
  _Float16* W2h    = (_Float16*)alloc((size_t)Hdim * Hdim * sizeof(_Float16));

  s4d_params<<<(Hdim * NMODE + 255) / 256, 256, 0, stream>>>(
      log_dt, log_A, A_im, C_re, C_im, params);
  cvt_weights<<<(Hdim * Hdim + 255) / 256, 256, 0, stream>>>(W1, W2, W1h, W2h);

  int nwaves = Bsz * Hdim * NCHUNK;                 // 131072 (b,h,chunk) waves
  scan_pass1<<<nwaves / 8, 256, 0, stream>>>(x, params, E);
  carry_combine<<<(Bsz * Hdim * NMODE + 255) / 256, 256, 0, stream>>>(params, E, carry);
  scan_pass2<<<nwaves / 8, 256, 0, stream>>>(x, params, carry, Dp, yconv);

  layernorm_f16<<<(Bsz * Lseq) / 8, 256, 0, stream>>>(yconv, ln_g, ln_b, yln);

  dim3 gg(Bsz * Lseq / 64, Hdim / 32);              // (1024, 8)
  gemm_relu<false><<<gg, 256, 0, stream>>>(yln, W1h, b1, nullptr, h1, nullptr);
  gemm_relu<true ><<<gg, 256, 0, stream>>>(h1, W2h, b2, x, nullptr, out);
}
